// BiRNN_41412074668072
// MI455X (gfx1250) — compile-verified, matrix operand order checked by reading, not run on hardware
//
#include <hip/hip_runtime.h>

#define BB   64    // batch
#define TT   512   // time
#define II   256   // input dim
#define HH   512   // hidden dim
#define CC   128   // classes

typedef __attribute__((ext_vector_type(16))) __bf16 v16bf;
typedef __attribute__((ext_vector_type(8)))  __bf16 v8bf;
typedef __attribute__((ext_vector_type(8)))  float  v8f;

// ---------- bf16 helpers (explicit RNE, no native casts) ----------
__device__ __forceinline__ __bf16 f2bf(float f) {
  unsigned int u = __builtin_bit_cast(unsigned int, f);
  u += 0x7fffu + ((u >> 16) & 1u);
  unsigned short h = (unsigned short)(u >> 16);
  return __builtin_bit_cast(__bf16, h);
}
__device__ __forceinline__ float bf2f(__bf16 v) {
  unsigned int x = ((unsigned int)__builtin_bit_cast(unsigned short, v)) << 16;
  return __builtin_bit_cast(float, x);
}

// ---------- WMMA fragment loads (16x32 bf16 A-layout; B mirrored) ----------
// lane 0-15: row = lane,    K elems {k0..k0+7} and {k0+16..k0+23}
// lane 16-31: row = lane-16, K elems {k0+8..k0+15} and {k0+24..k0+31}
__device__ __forceinline__ v16bf load_frag(const __bf16* tile, int stride, int lane) {
  const __bf16* p = tile + (lane & 15) * stride + ((lane >> 4) << 3);
  v8bf lo = *(const v8bf*)(p);
  v8bf hi = *(const v8bf*)(p + 16);
  v16bf r;
#pragma unroll
  for (int i = 0; i < 8; ++i) { r[i] = lo[i]; r[i + 8] = hi[i]; }
  return r;
}
__device__ __forceinline__ v16bf load_frag_f32(const float* tile, int stride, int lane) {
  const float* p = tile + (lane & 15) * stride + ((lane >> 4) << 3);
  v16bf r;
#pragma unroll
  for (int i = 0; i < 8; ++i) { r[i] = f2bf(p[i]); r[i + 8] = f2bf(p[i + 16]); }
  return r;
}
__device__ __forceinline__ v8f wmma_bf16(v16bf a, v16bf b, v8f c) {
  return __builtin_amdgcn_wmma_f32_16x16x32_bf16(false, a, false, b, (short)0, c,
                                                 false, false);
}

// ---------- kernel 0: f32 -> bf16 convert ----------
__global__ void cvt_f32_bf16(const float* __restrict__ src,
                             __bf16* __restrict__ dst, int n) {
  int i = blockIdx.x * blockDim.x + threadIdx.x;
  if (i < n) dst[i] = f2bf(src[i]);
}

// ---------- kernel 1: input projection GEMM ----------
// out[m, n] = sum_k x[b, t, k] * Wx[n, k] + bx[n], m = t*BB + b, stored [T*B, H] bf16
__global__ __launch_bounds__(256) void proj_kernel(
    const __bf16* __restrict__ xb16,
    const __bf16* __restrict__ wxf, const __bf16* __restrict__ wxb,
    const float* __restrict__ bxf, const float* __restrict__ bxb,
    __bf16* __restrict__ xwf, __bf16* __restrict__ xwb) {
  const int lane = threadIdx.x & 31;
  const int wv = threadIdx.x >> 5;           // 0..7
  const int wm = wv & 1, wn = wv >> 1;       // 2 (M) x 4 (N) wave grid
  const int dir = blockIdx.z;
  const __bf16* W = dir ? wxb : wxf;
  const float* bias = dir ? bxb : bxf;
  __bf16* out = dir ? xwb : xwf;
  const int m0 = blockIdx.x * 128 + wm * 64; // wave M base (of 32768)
  const int n0 = blockIdx.y * 128 + wn * 32; // wave N base (of 512)

  v8f acc[4][2] = {};
#pragma unroll
  for (int kk = 0; kk < II; kk += 32) {
    v16bf Bf0 = load_frag(W + (size_t)n0 * II + kk, II, lane);
    v16bf Bf1 = load_frag(W + (size_t)(n0 + 16) * II + kk, II, lane);
#pragma unroll
    for (int mt = 0; mt < 4; ++mt) {
      int m = m0 + mt * 16;
      int b0 = m & (BB - 1), t = m >> 6;     // BB == 64
      v16bf Af = load_frag(xb16 + ((size_t)b0 * TT + t) * II + kk, TT * II, lane);
      acc[mt][0] = wmma_bf16(Af, Bf0, acc[mt][0]);
      acc[mt][1] = wmma_bf16(Af, Bf1, acc[mt][1]);
    }
  }
  const int col = lane & 15;
  const int rhi = (lane >> 4) << 3;
#pragma unroll
  for (int mt = 0; mt < 4; ++mt)
#pragma unroll
    for (int nt = 0; nt < 2; ++nt) {
      int n = n0 + nt * 16 + col;
      float bv = bias[n];
#pragma unroll
      for (int v = 0; v < 8; ++v) {
        int m = m0 + mt * 16 + v + rhi;
        out[(size_t)m * HH + n] = f2bf(acc[mt][nt][v] + bv);
      }
    }
}

// ---------- kernel 2: persistent bidirectional scan ----------
// One workgroup per direction; h[64][512] in LDS; wave w owns 16 output cols;
// its Wh slice (16 B-fragments, 128 VGPRs) stays register-resident for all T steps.
#define HPAD 520
__global__ __launch_bounds__(1024) void scan_kernel(
    const __bf16* __restrict__ xwf, const __bf16* __restrict__ xwb,
    const __bf16* __restrict__ whf, const __bf16* __restrict__ whb,
    const float* __restrict__ bhf, const float* __restrict__ bhb,
    float* __restrict__ hcat) {
  extern __shared__ __bf16 hshm[];  // [64][HPAD] = 66,560 bytes
  const int dir = blockIdx.x;
  const __bf16* xw = dir ? xwb : xwf;
  const __bf16* Wh = dir ? whb : whf;
  const float* bh = dir ? bhb : bhf;
  const int lane = threadIdx.x & 31;
  const int w = threadIdx.x >> 5;   // 0..31 -> N-tile
  const int col = lane & 15;
  const int rhi = (lane >> 4) << 3;
  const int n = w * 16 + col;
  const float bias = bh[n];

  for (int i = threadIdx.x; i < BB * HPAD; i += 1024)
    hshm[i] = __builtin_bit_cast(__bf16, (unsigned short)0);

  v16bf Bf[16];                     // Wh[w*16 .. w*16+15][:] as 16 K-fragments
#pragma unroll
  for (int kt = 0; kt < 16; ++kt)
    Bf[kt] = load_frag(Wh + (size_t)(w * 16) * HH + kt * 32, HH, lane);

  __syncthreads();
  for (int t = 0; t < TT; ++t) {
    const int tt = dir ? (TT - 1 - t) : t;
    const __bf16* xt = xw + (size_t)tt * (BB * HH);
    if (t + 1 < TT) {
      const int tn = dir ? (TT - 2 - t) : (t + 1);
      __builtin_prefetch(xw + (size_t)tn * (BB * HH) + threadIdx.x * 32, 0, 0);
    }
    v8f acc[4] = {};
#pragma unroll
    for (int kt = 0; kt < 16; ++kt) {
#pragma unroll
      for (int mt = 0; mt < 4; ++mt) {
        v16bf Af = load_frag(&hshm[(mt * 16) * HPAD + kt * 32], HPAD, lane);
        acc[mt] = wmma_bf16(Af, Bf[kt], acc[mt]);
      }
    }
    __syncthreads();  // all waves done reading old h
#pragma unroll
    for (int mt = 0; mt < 4; ++mt) {
#pragma unroll
      for (int v = 0; v < 8; ++v) {
        int b = mt * 16 + v + rhi;
        float val = acc[mt][v] + bf2f(xt[(size_t)b * HH + n]) + bias;
        float hv = tanhf(val);
        hshm[b * HPAD + n] = f2bf(hv);
        if (t == TT - 1) hcat[(size_t)b * (2 * HH) + dir * HH + n] = hv;
      }
    }
    __syncthreads();  // new h visible before next step's reads
  }
}

// ---------- kernel 3: final FC ----------
__global__ __launch_bounds__(256) void fc_kernel(
    const float* __restrict__ hcat, const __bf16* __restrict__ fcw,
    const float* __restrict__ fcb, float* __restrict__ out) {
  const int lane = threadIdx.x & 31;
  const int w = threadIdx.x >> 5;   // 0..7 -> N-tile of C=128
  v8f acc[4] = {};
  for (int kt = 0; kt < (2 * HH) / 32; ++kt) {
    v16bf Bf = load_frag(fcw + (size_t)(w * 16) * (2 * HH) + kt * 32, 2 * HH, lane);
#pragma unroll
    for (int mt = 0; mt < 4; ++mt) {
      v16bf Af = load_frag_f32(hcat + (size_t)(mt * 16) * (2 * HH) + kt * 32,
                               2 * HH, lane);
      acc[mt] = wmma_bf16(Af, Bf, acc[mt]);
    }
  }
  const int c = w * 16 + (lane & 15);
  const int rhi = (lane >> 4) << 3;
  const float bv = fcb[c];
#pragma unroll
  for (int mt = 0; mt < 4; ++mt)
#pragma unroll
    for (int v = 0; v < 8; ++v) {
      int b = mt * 16 + v + rhi;
      out[(size_t)b * CC + c] = acc[mt][v] + bv;
    }
}

extern "C" void kernel_launch(void* const* d_in, const int* in_sizes, int n_in,
                              void* d_out, int out_size, void* d_ws, size_t ws_size,
                              hipStream_t stream) {
  (void)in_sizes; (void)n_in; (void)out_size; (void)ws_size;
  const float* x     = (const float*)d_in[0];
  const float* Wxf_w = (const float*)d_in[1];
  const float* Wxf_b = (const float*)d_in[2];
  const float* Whf_w = (const float*)d_in[3];
  const float* Whf_b = (const float*)d_in[4];
  const float* Wxb_w = (const float*)d_in[5];
  const float* Wxb_b = (const float*)d_in[6];
  const float* Whb_w = (const float*)d_in[7];
  const float* Whb_b = (const float*)d_in[8];
  const float* fc_w  = (const float*)d_in[9];
  const float* fc_b  = (const float*)d_in[10];
  float* out = (float*)d_out;

  char* ws = (char*)d_ws;
  size_t off = 0;
  __bf16* xb16  = (__bf16*)(ws + off); off += (size_t)BB * TT * II * 2;
  __bf16* wxf16 = (__bf16*)(ws + off); off += (size_t)HH * II * 2;
  __bf16* wxb16 = (__bf16*)(ws + off); off += (size_t)HH * II * 2;
  __bf16* whf16 = (__bf16*)(ws + off); off += (size_t)HH * HH * 2;
  __bf16* whb16 = (__bf16*)(ws + off); off += (size_t)HH * HH * 2;
  __bf16* fcw16 = (__bf16*)(ws + off); off += (size_t)CC * 2 * HH * 2;
  __bf16* xwf   = (__bf16*)(ws + off); off += (size_t)TT * BB * HH * 2;
  __bf16* xwb   = (__bf16*)(ws + off); off += (size_t)TT * BB * HH * 2;
  float*  hcat  = (float*)(ws + off);  off += (size_t)BB * 2 * HH * 4;

  // 0) convert inputs/weights to bf16
  {
    int n;
    n = BB * TT * II;  cvt_f32_bf16<<<(n + 255) / 256, 256, 0, stream>>>(x, xb16, n);
    n = HH * II;       cvt_f32_bf16<<<(n + 255) / 256, 256, 0, stream>>>(Wxf_w, wxf16, n);
    n = HH * II;       cvt_f32_bf16<<<(n + 255) / 256, 256, 0, stream>>>(Wxb_w, wxb16, n);
    n = HH * HH;       cvt_f32_bf16<<<(n + 255) / 256, 256, 0, stream>>>(Whf_w, whf16, n);
    n = HH * HH;       cvt_f32_bf16<<<(n + 255) / 256, 256, 0, stream>>>(Whb_w, whb16, n);
    n = CC * 2 * HH;   cvt_f32_bf16<<<(n + 255) / 256, 256, 0, stream>>>(fc_w, fcw16, n);
  }
  // 1) input projections, both directions: [T*B, H]
  proj_kernel<<<dim3((TT * BB) / 128, HH / 128, 2), 256, 0, stream>>>(
      xb16, wxf16, wxb16, Wxf_b, Wxb_b, xwf, xwb);
  // 2) persistent scans (1 workgroup per direction, 32 waves, Wh in registers)
  scan_kernel<<<2, 1024, BB * HPAD * 2, stream>>>(
      xwf, xwb, whf16, whb16, Whf_b, Whb_b, hcat);
  // 3) final FC
  fc_kernel<<<1, 256, 0, stream>>>(hcat, fcw16, fc_b, out);
}
